// MyQKVAttention_19035295056227
// MI455X (gfx1250) — matrix operand order repaired and checked by reading, not compiled, not run on hardware
//
#include <hip/hip_runtime.h>

typedef __attribute__((ext_vector_type(16))) _Float16 v16h;
typedef __attribute__((ext_vector_type(8)))  _Float16 v8h;
typedef __attribute__((ext_vector_type(4)))  _Float16 v4h;
typedef __attribute__((ext_vector_type(8)))  float    v8f;

#define DIMC 64
#define SEQ  4096
#define BQ   128   // queries per block (8 waves x 16)
#define TN   64    // keys per iteration
#define NIT  (SEQ / TN)

union V16U { v16h v; v8h h[2]; };

// ---- DPP16 butterfly reductions over each 16-lane group (no LDS) ----
template <int CTRL>
__device__ __forceinline__ float dpp_mov(float x) {
    int xi = __builtin_bit_cast(int, x);
    int r  = __builtin_amdgcn_update_dpp(xi, xi, CTRL, 0xF, 0xF, true);
    return __builtin_bit_cast(float, r);
}
__device__ __forceinline__ float rowmax16(float x) {
    x = fmaxf(x, dpp_mov<0xB1>(x));  // quad_perm [1,0,3,2]  : xor 1
    x = fmaxf(x, dpp_mov<0x4E>(x));  // quad_perm [2,3,0,1]  : xor 2
    x = fmaxf(x, dpp_mov<0x141>(x)); // row_half_mirror      : combines quads
    x = fmaxf(x, dpp_mov<0x140>(x)); // row_mirror           : combines halves
    return x;
}
__device__ __forceinline__ float rowsum16(float x) {
    x += dpp_mov<0xB1>(x);
    x += dpp_mov<0x4E>(x);
    x += dpp_mov<0x141>(x);
    x += dpp_mov<0x140>(x);
    return x;
}

__global__ __launch_bounds__(256)
void qkv_flash_attn_f16wmma(const float* __restrict__ qkv, float* __restrict__ out)
{
    __shared__ __align__(16) _Float16 Qs[BQ * DIMC];        // [s][c], q * (0.125*log2e)
    __shared__ __align__(16) _Float16 Kbuf[2][TN * DIMC];   // [t][c], double-buffered
    __shared__ __align__(16) _Float16 Vbuf[2][DIMC * TN];   // [d][t], double-buffered
    __shared__ __align__(16) _Float16 Pw[8 * 16 * TN];      // per-wave P tile [s][t]

    const int tid  = threadIdx.x;
    const int wave = tid >> 5;
    const int lane = tid & 31;
    const int col  = lane & 15;   // column/row index inside a 16-lane group
    const int hi   = lane >> 4;   // which 16-lane half

    const int b     = blockIdx.y;
    const int sBase = blockIdx.x * BQ;

    const float* qb = qkv + (size_t)b * (3 * DIMC) * SEQ;
    const float* kb = qb + (size_t)DIMC * SEQ;
    const float* vb = qb + (size_t)(2 * DIMC) * SEQ;

    // (1/sqrt(sqrt(64)))^2 = 1/8 folded into q, plus log2(e) for exp2-based softmax
    const float QSCALE = 0.125f * 1.4426950408889634f;

    // ---- stage Q block to LDS (coalesced in s, transposed to [s][c]) ----
    for (int idx = tid; idx < BQ * DIMC; idx += 256) {
        int s = idx & (BQ - 1);
        int c = idx >> 7;
        Qs[s * DIMC + c] = (_Float16)(qb[c * SEQ + sBase + s] * QSCALE);
    }

    // ---- K/V tile pipeline: global -> regs -> (f16) LDS, double-buffered ----
    const int pf_t = tid & 63;
    const int pf_c = tid >> 6;
    float kreg[16], vreg[16];

    auto loadTile = [&](int tb) {
        const float* kn = kb + tb + pf_t;
        const float* vn = vb + tb + pf_t;
#pragma unroll
        for (int k = 0; k < 16; ++k) {
            int c = pf_c + 4 * k;
            kreg[k] = kn[c * SEQ];
            vreg[k] = vn[c * SEQ];
        }
    };
    auto stageTile = [&](_Float16* K_, _Float16* V_) {
#pragma unroll
        for (int k = 0; k < 16; ++k) {
            int c = pf_c + 4 * k;
            K_[pf_t * DIMC + c] = (_Float16)kreg[k];  // transpose to [t][c]
            V_[c * TN + pf_t]   = (_Float16)vreg[k];  // direct copy [d][t]
        }
    };

    loadTile(0);
    stageTile(Kbuf[0], Vbuf[0]);
    loadTile(TN);
    __syncthreads();   // Qs + first K/V tile visible to all waves

    // ---- per-wave A fragments of Q^T: M = 16 queries, K = c (2 chunks of 32) ----
    // 16-bit A layout: lanes<16 hold K 0..7 / 16..23 ; lanes>=16 hold K 8..15 / 24..31
    const v8h* qrow = (const v8h*)&Qs[(wave * 16 + col) * DIMC];
    V16U a0u, a1u;
    a0u.h[0] = qrow[hi];     a0u.h[1] = qrow[2 + hi];
    a1u.h[0] = qrow[4 + hi]; a1u.h[1] = qrow[6 + hi];
    const v16h a0 = a0u.v, a1 = a1u.v;

    v8f   Oacc[4] = {};
    float mrun[8], lrun[8];
#pragma unroll
    for (int j = 0; j < 8; ++j) { mrun[j] = -1e30f; lrun[j] = 0.0f; }

    _Float16* PwW = &Pw[wave * 16 * TN];

    for (int it = 0; it < NIT; ++it) {
        const _Float16* Kt = Kbuf[it & 1];
        const _Float16* Vt = Vbuf[it & 1];

        // stage tile it+1 into the idle buffer (overlaps with compute below),
        // then kick off global loads for tile it+2 (LOADcnt outstanding through compute)
        if (it + 1 < NIT) stageTile(Kbuf[(it + 1) & 1], Vbuf[(it + 1) & 1]);
        if (it + 2 < NIT) loadTile((it + 2) * TN);

        // ---- S = Q^T K : 4 subtiles; subtile n covers keys t = 4*col + n ----
        v8f Sc[4];
#pragma unroll
        for (int n = 0; n < 4; ++n) {
            const v8h* krow = (const v8h*)&Kt[(col * 4 + n) * DIMC];
            V16U b0u, b1u;   // B layout: lanes<16 hold K 0..7 / 16..23 ; lanes>=16 hold K 8..15 / 24..31
            b0u.h[0] = krow[hi];     b0u.h[1] = krow[2 + hi];
            b1u.h[0] = krow[4 + hi]; b1u.h[1] = krow[6 + hi];
            v8f acc = {};
            acc = __builtin_amdgcn_wmma_f32_16x16x32_f16(false, a1, false, b1u.v,
                                                         (short)0, acc, false, false);
            acc = __builtin_amdgcn_wmma_f32_16x16x32_f16(false, a0, false, b0u.v,
                                                         (short)0, acc, false, false);
            Sc[n] = acc;
        }

        // ---- online softmax (base-2); row j lives in lanes {0..15} (M=j) or {16..31} (M=j+8)
#pragma unroll
        for (int j = 0; j < 8; ++j) {
            float m = fmaxf(fmaxf(Sc[0][j], Sc[1][j]), fmaxf(Sc[2][j], Sc[3][j]));
            m = rowmax16(m);
            float mn = fmaxf(mrun[j], m);
            float cj = __builtin_amdgcn_exp2f(mrun[j] - mn);
            mrun[j] = mn;
            lrun[j] *= cj;
#pragma unroll
            for (int d = 0; d < 4; ++d) Oacc[d][j] *= cj;

            int r = hi * 8 + j;
            float rs = 0.0f;
            v4h ph;
#pragma unroll
            for (int n = 0; n < 4; ++n) {
                float p = __builtin_amdgcn_exp2f(Sc[n][j] - mn);
                rs += p;
                ph[n] = (_Float16)p;            // keys 4*col + 0..3 are adjacent
            }
            *(v4h*)&PwW[r * TN + col * 4] = ph; // one ds_store_b64 per row
            lrun[j] += rowsum16(rs);
        }

        // per-wave LDS buffer: only need DS ops complete (DS in-order within a wave)
        asm volatile("s_wait_dscnt 0" ::: "memory");

        // ---- read P back in A layout (M=s, K=t, 2 chunks of 32) ----
        const v8h* prow = (const v8h*)&PwW[col * TN];
        V16U p0u, p1u;
        p0u.h[0] = prow[hi];     p0u.h[1] = prow[2 + hi];
        p1u.h[0] = prow[4 + hi]; p1u.h[1] = prow[6 + hi];

        // ---- O += P x V' : 4 d-subtiles x 2 K-chunks ----
#pragma unroll
        for (int dsub = 0; dsub < 4; ++dsub) {
            const v8h* vrow = (const v8h*)&Vt[(dsub * 16 + col) * TN];
            V16U bv0, bv1;
            bv0.h[0] = vrow[hi];     bv0.h[1] = vrow[2 + hi];
            bv1.h[0] = vrow[4 + hi]; bv1.h[1] = vrow[6 + hi];
            Oacc[dsub] = __builtin_amdgcn_wmma_f32_16x16x32_f16(false, p1u.v, false, bv1.v,
                                                                (short)0, Oacc[dsub], false, false);
            Oacc[dsub] = __builtin_amdgcn_wmma_f32_16x16x32_f16(false, p0u.v, false, bv0.v,
                                                                (short)0, Oacc[dsub], false, false);
        }

        __syncthreads();   // next tile fully staged; current tile readers done
    }

    // ---- epilogue: O /= l, store out[b][d][s]; rows j map to consecutive s ----
    float linv[8];
#pragma unroll
    for (int j = 0; j < 8; ++j) linv[j] = 1.0f / lrun[j];

    float* ob = out + (size_t)b * DIMC * SEQ;
    const int sOut = sBase + wave * 16 + hi * 8;
#pragma unroll
    for (int dsub = 0; dsub < 4; ++dsub) {
        int d = dsub * 16 + col;
        float4 lo, hi4;
        lo.x  = Oacc[dsub][0] * linv[0]; lo.y  = Oacc[dsub][1] * linv[1];
        lo.z  = Oacc[dsub][2] * linv[2]; lo.w  = Oacc[dsub][3] * linv[3];
        hi4.x = Oacc[dsub][4] * linv[4]; hi4.y = Oacc[dsub][5] * linv[5];
        hi4.z = Oacc[dsub][6] * linv[6]; hi4.w = Oacc[dsub][7] * linv[7];
        *(float4*)&ob[(size_t)d * SEQ + sOut]     = lo;
        *(float4*)&ob[(size_t)d * SEQ + sOut + 4] = hi4;
    }
}

extern "C" void kernel_launch(void* const* d_in, const int* in_sizes, int n_in,
                              void* d_out, int out_size, void* d_ws, size_t ws_size,
                              hipStream_t stream)
{
    const float* qkv = (const float*)d_in[0];
    float* out = (float*)d_out;
    dim3 grid(SEQ / BQ, 8, 1);
    qkv_flash_attn_f16wmma<<<grid, 256, 0, stream>>>(qkv, out);
}
